// AttnBlock_21131239097052
// MI455X (gfx1250) — matrix operand order repaired and checked by reading, not compiled, not run on hardware
//
#include <hip/hip_runtime.h>
#include <hip/hip_bf16.h>
#include <math.h>

typedef __attribute__((ext_vector_type(16))) _Float16 v16h;
typedef __attribute__((ext_vector_type(8)))  _Float16 v8h;
typedef __attribute__((ext_vector_type(8)))  float    v8f;

#define NB   8      // batch
#define NC   256    // channels
#define NG   32     // groups
#define NHW  1024   // H*W
#define EPSV 1e-5f

// ---------------------------------------------------------------------------
// WMMA helpers (CDNA5 wave32, v_wmma_f32_16x16x32_f16)
// ---------------------------------------------------------------------------
__device__ __forceinline__ v8f wmma16(v16h a, v16h b, v8f c) {
  return __builtin_amdgcn_wmma_f32_16x16x32_f16(
      false, a, false, b, (short)0, c, false, false);
}

// A fragment (16x32 f16, M=lane%16): K=e+16*hi
__device__ __forceinline__ v16h load_a_h(const _Float16* row, int hi) {
  const v8h lo = *(const v8h*)(row + (hi ? 16 : 0));
  const v8h hh = *(const v8h*)(row + (hi ? 24 : 8));
  return __builtin_shufflevector(lo, hh, 0,1,2,3,4,5,6,7,8,9,10,11,12,13,14,15);
}

// B fragment (32x16 f16, N=lane%16, K=e+16*hi): 16 contiguous halves at col base
__device__ __forceinline__ v16h load_b_h(const _Float16* col) {
  return *(const v16h*)col;
}

// ---------------------------------------------------------------------------
// Kernel 0: one-shot f32 -> f16 weight conversion (n = 2*C*C per tensor)
// ---------------------------------------------------------------------------
__global__ void k_cvt_w(const float* __restrict__ src, _Float16* __restrict__ dst) {
  const int i = (blockIdx.x * 256 + threadIdx.x) * 4;
#pragma unroll
  for (int k = 0; k < 4; ++k) dst[i + k] = (_Float16)src[i + k];
}

// ---------------------------------------------------------------------------
// Kernel 1: GroupNorm (real & imag separately) -> f16, pixel-major [p,b,hw,c]
// grid = 2*8*32 blocks of 256
// ---------------------------------------------------------------------------
__global__ void k_groupnorm(const float* __restrict__ x2,
                            const float* __restrict__ gw,
                            const float* __restrict__ gb,
                            _Float16* __restrict__ h) {
  __shared__ float s1[256];
  __shared__ float s2[256];
  const int blk = blockIdx.x;
  const int p   = blk >> 8;
  const int rem = blk & 255;
  const int b   = rem >> 5;
  const int g   = rem & 31;
  const int t   = threadIdx.x;
  const float* xg = x2 + (((size_t)p * NB + b) * NC + g * 8) * NHW;

  float vals[32];
  float sum = 0.f, ss = 0.f;
#pragma unroll
  for (int k = 0; k < 32; ++k) {
    float xv = xg[t + 256 * k];
    vals[k] = xv;
    sum += xv;
    ss  += xv * xv;
  }
  s1[t] = sum; s2[t] = ss;
  __syncthreads();
  for (int s = 128; s > 0; s >>= 1) {
    if (t < s) { s1[t] += s1[t + s]; s2[t] += s2[t + s]; }
    __syncthreads();
  }
  const float mean = s1[0] * (1.0f / 8192.0f);
  const float var  = s2[0] * (1.0f / 8192.0f) - mean * mean;
  const float rstd = rsqrtf(var + EPSV);

  _Float16* hb = h + (((size_t)p * NB + b) * NHW) * NC;
#pragma unroll
  for (int k = 0; k < 32; ++k) {
    int e  = t + 256 * k;
    int cl = e >> 10;
    int hw = e & 1023;
    int c  = g * 8 + cl;
    float v = (vals[k] - mean) * rstd * gw[p * NC + c] + gb[p * NC + c];
    hb[(size_t)hw * NC + c] = (_Float16)v;
  }
}

// ---------------------------------------------------------------------------
// Kernel 2: complex 1x1-conv projection  out = scale*(Wc*h + bc)
// 16x32 tile per wave (A fragments reused across two N-tiles).
// mode 0: store pixel-major [p,b,hw,c]; mode 1: channel-major [p,b,c,hw]
// grid = 512 blocks x 256 (4096 waves = 8b x 16m x 32n-pairs)
// ---------------------------------------------------------------------------
__global__ void k_proj(const _Float16* __restrict__ hin,
                       const _Float16* __restrict__ w16,  // [2,C,C] f16
                       const float* __restrict__ b2,      // [2,C]
                       _Float16* __restrict__ outp,
                       int mode, float scale) {
  const int lane = threadIdx.x & 31;
  const int wid  = blockIdx.x * 8 + (threadIdx.x >> 5);  // 0..4095
  const int b    = wid >> 9;
  const int rem  = wid & 511;
  const int m0   = (rem >> 5) * 16;   // c_out tile
  const int n0   = (rem & 31) * 32;   // hw double-tile
  const int hi   = lane >> 4, lm = lane & 15;

  const _Float16* wrRow = w16 + (size_t)(m0 + lm) * NC;
  const _Float16* wiRow = w16 + (size_t)NC * NC + (size_t)(m0 + lm) * NC;
  const _Float16* hr0 = hin + (((size_t)b) * NHW + n0 + lm) * NC + 16 * hi;
  const _Float16* hi0 = hin + (((size_t)NB + b) * NHW + n0 + lm) * NC + 16 * hi;
  const _Float16* hr1 = hr0 + (size_t)16 * NC;
  const _Float16* hi1 = hi0 + (size_t)16 * NC;

  v8f ar0, ai0, ar1, ai1;
  const int mRow = m0 + 8 * hi;
#pragma unroll
  for (int r = 0; r < 8; ++r) {
    float br = b2[mRow + r], bi = b2[NC + mRow + r];
    ar0[r] = br; ar1[r] = br; ai0[r] = bi; ai1[r] = bi;
  }

#pragma unroll 2
  for (int kk = 0; kk < NC; kk += 32) {
    if (kk + 32 < NC) {
      __builtin_prefetch(hr0 + kk + 32, 0, 3);
      __builtin_prefetch(hr1 + kk + 32, 0, 3);
    }
    v16h fwr = load_a_h(wrRow + kk, hi);
    v16h fwi = load_a_h(wiRow + kk, hi);
    v16h fwn = -fwi;                       // -Wi (f16 WMMA has no A-neg)
    v16h fhr0 = load_b_h(hr0 + kk);
    v16h fhi0 = load_b_h(hi0 + kk);
    v16h fhr1 = load_b_h(hr1 + kk);
    v16h fhi1 = load_b_h(hi1 + kk);
    ar0 = wmma16(fwr, fhr0, ar0);  ar0 = wmma16(fwn, fhi0, ar0);
    ai0 = wmma16(fwr, fhi0, ai0);  ai0 = wmma16(fwi, fhr0, ai0);
    ar1 = wmma16(fwr, fhr1, ar1);  ar1 = wmma16(fwn, fhi1, ar1);
    ai1 = wmma16(fwr, fhi1, ai1);  ai1 = wmma16(fwi, fhr1, ai1);
  }

#pragma unroll
  for (int r = 0; r < 8; ++r) {
    const int M  = mRow + r;          // c_out
    const int N0 = n0 + lm;           // hw (tile 0)
    const int N1 = n0 + 16 + lm;      // hw (tile 1)
    if (mode == 0) {
      outp[(((size_t)b) * NHW + N0) * NC + M]      = (_Float16)(ar0[r] * scale);
      outp[(((size_t)NB + b) * NHW + N0) * NC + M] = (_Float16)(ai0[r] * scale);
      outp[(((size_t)b) * NHW + N1) * NC + M]      = (_Float16)(ar1[r] * scale);
      outp[(((size_t)NB + b) * NHW + N1) * NC + M] = (_Float16)(ai1[r] * scale);
    } else {
      outp[(((size_t)b) * NC + M) * NHW + N0]      = (_Float16)(ar0[r] * scale);
      outp[(((size_t)NB + b) * NC + M) * NHW + N0] = (_Float16)(ai0[r] * scale);
      outp[(((size_t)b) * NC + M) * NHW + N1]      = (_Float16)(ar1[r] * scale);
      outp[(((size_t)NB + b) * NC + M) * NHW + N1] = (_Float16)(ai1[r] * scale);
    }
  }
}

// ---------------------------------------------------------------------------
// Kernel 3: scores S[b,i,j] = qr.kr + qi.ki  (scale folded into q), f32 out
// 16x32 tile per wave. grid = 2048 blocks (16384 waves = 8b x 64i x 32j-pairs)
// ---------------------------------------------------------------------------
__global__ void k_scores(const _Float16* __restrict__ q,
                         const _Float16* __restrict__ kf,
                         float* __restrict__ S) {
  const int lane = threadIdx.x & 31;
  const int wid  = blockIdx.x * 8 + (threadIdx.x >> 5);  // 0..16383
  const int b    = wid >> 11;
  const int rem  = wid & 2047;
  const int i0   = (rem >> 5) * 16;
  const int j0   = (rem & 31) * 32;
  const int hi   = lane >> 4, lm = lane & 15;

  const _Float16* qrR = q  + (((size_t)b) * NHW + i0 + lm) * NC;
  const _Float16* qiR = q  + (((size_t)NB + b) * NHW + i0 + lm) * NC;
  const _Float16* kr0 = kf + (((size_t)b) * NHW + j0 + lm) * NC + 16 * hi;
  const _Float16* ki0 = kf + (((size_t)NB + b) * NHW + j0 + lm) * NC + 16 * hi;
  const _Float16* kr1 = kr0 + (size_t)16 * NC;
  const _Float16* ki1 = ki0 + (size_t)16 * NC;

  v8f acc0 = {}, acc1 = {};
#pragma unroll 2
  for (int c0 = 0; c0 < NC; c0 += 32) {
    v16h aqr = load_a_h(qrR + c0, hi);
    v16h aqi = load_a_h(qiR + c0, hi);
    v16h b0r = load_b_h(kr0 + c0);
    v16h b0i = load_b_h(ki0 + c0);
    v16h b1r = load_b_h(kr1 + c0);
    v16h b1i = load_b_h(ki1 + c0);
    acc0 = wmma16(aqr, b0r, acc0);  acc0 = wmma16(aqi, b0i, acc0);
    acc1 = wmma16(aqr, b1r, acc1);  acc1 = wmma16(aqi, b1i, acc1);
  }
  float* srow = S + ((size_t)b * NHW + i0 + 8 * hi) * NHW + j0 + lm;
#pragma unroll
  for (int r = 0; r < 8; ++r) {
    srow[(size_t)r * NHW]      = acc0[r];
    srow[(size_t)r * NHW + 16] = acc1[r];
  }
}

// ---------------------------------------------------------------------------
// Kernel 4: row softmax over 1024; reads f32 S, writes f16 A (row-major).
// grid = 8192 x 256
// ---------------------------------------------------------------------------
__global__ void k_softmax(const float* __restrict__ S, _Float16* __restrict__ A16) {
  __shared__ float red[256];
  const int t = threadIdx.x;
  const float* p = S + (size_t)blockIdx.x * NHW;
  _Float16* po = A16 + (size_t)blockIdx.x * NHW;
  float v[4];
  float m = -3.4e38f;
#pragma unroll
  for (int k = 0; k < 4; ++k) { v[k] = p[t + 256 * k]; m = fmaxf(m, v[k]); }
  red[t] = m; __syncthreads();
  for (int s = 128; s > 0; s >>= 1) {
    if (t < s) red[t] = fmaxf(red[t], red[t + s]);
    __syncthreads();
  }
  m = red[0];
  __syncthreads();
  float sum = 0.f;
#pragma unroll
  for (int k = 0; k < 4; ++k) { v[k] = __expf(v[k] - m); sum += v[k]; }
  red[t] = sum; __syncthreads();
  for (int s = 128; s > 0; s >>= 1) {
    if (t < s) red[t] += red[t + s];
    __syncthreads();
  }
  const float inv = 1.0f / red[0];
#pragma unroll
  for (int k = 0; k < 4; ++k) po[t + 256 * k] = (_Float16)(v[k] * inv);
}

// ---------------------------------------------------------------------------
// Kernel 5: O = A @ V (A real f16 row-major, V complex f16 channel-major)
// 16x32 tile per wave, A fragment shared across 4 WMMAs.
// grid = 512 blocks (4096 waves = 8b x 64i x 8c-pairs)
// ---------------------------------------------------------------------------
__global__ void k_av(const _Float16* __restrict__ A16,
                     const _Float16* __restrict__ vbuf,
                     _Float16* __restrict__ obuf) {
  const int lane = threadIdx.x & 31;
  const int wid  = blockIdx.x * 8 + (threadIdx.x >> 5);  // 0..4095
  const int b    = wid >> 9;
  const int rem  = wid & 511;
  const int i0   = (rem >> 3) * 16;   // hw tile
  const int c0   = (rem & 7) * 32;    // channel double-tile
  const int hi   = lane >> 4, lm = lane & 15;

  const _Float16* arow = A16 + ((size_t)b * NHW + i0 + lm) * NHW;
  const _Float16* vr0  = vbuf + (((size_t)b) * NC + c0 + lm) * NHW + 16 * hi;
  const _Float16* vi0  = vbuf + (((size_t)NB + b) * NC + c0 + lm) * NHW + 16 * hi;
  const _Float16* vr1  = vr0 + (size_t)16 * NHW;
  const _Float16* vi1  = vi0 + (size_t)16 * NHW;

  v8f ar0 = {}, ai0 = {}, ar1 = {}, ai1 = {};
#pragma unroll 4
  for (int j = 0; j < NHW; j += 32) {
    if (j + 32 < NHW) {
      __builtin_prefetch(arow + j + 32, 0, 3);
      __builtin_prefetch(vr0 + j + 32, 0, 3);
      __builtin_prefetch(vr1 + j + 32, 0, 3);
    }
    v16h af  = load_a_h(arow + j, hi);
    v16h b0r = load_b_h(vr0 + j);
    v16h b0i = load_b_h(vi0 + j);
    v16h b1r = load_b_h(vr1 + j);
    v16h b1i = load_b_h(vi1 + j);
    ar0 = wmma16(af, b0r, ar0);
    ai0 = wmma16(af, b0i, ai0);
    ar1 = wmma16(af, b1r, ar1);
    ai1 = wmma16(af, b1i, ai1);
  }
#pragma unroll
  for (int r = 0; r < 8; ++r) {
    const int M  = i0 + 8 * hi + r;   // hw
    const int N0 = c0 + lm;           // channel tile 0
    const int N1 = c0 + 16 + lm;      // channel tile 1
    obuf[(((size_t)b) * NHW + M) * NC + N0]      = (_Float16)ar0[r];
    obuf[(((size_t)NB + b) * NHW + M) * NC + N0] = (_Float16)ai0[r];
    obuf[(((size_t)b) * NHW + M) * NC + N1]      = (_Float16)ar1[r];
    obuf[(((size_t)NB + b) * NHW + M) * NC + N1] = (_Float16)ai1[r];
  }
}

// ---------------------------------------------------------------------------
// Kernel 6: out = x + Wo*o + bo  (complex), output f32 [2,B,C,HW]
// 16x32 tile per wave. grid = 512 blocks (4096 waves)
// ---------------------------------------------------------------------------
__global__ void k_final(const _Float16* __restrict__ obuf,
                        const _Float16* __restrict__ wo16,  // [2,C,C] f16
                        const float* __restrict__ bo,       // [2,C]
                        const float* __restrict__ x2,
                        float* __restrict__ outp) {
  const int lane = threadIdx.x & 31;
  const int wid  = blockIdx.x * 8 + (threadIdx.x >> 5);  // 0..4095
  const int b    = wid >> 9;
  const int rem  = wid & 511;
  const int m0   = (rem >> 5) * 16;   // c_out tile
  const int n0   = (rem & 31) * 32;   // hw double-tile
  const int hi   = lane >> 4, lm = lane & 15;

  const _Float16* wrRow = wo16 + (size_t)(m0 + lm) * NC;
  const _Float16* wiRow = wo16 + (size_t)NC * NC + (size_t)(m0 + lm) * NC;
  const _Float16* or0 = obuf + (((size_t)b) * NHW + n0 + lm) * NC + 16 * hi;
  const _Float16* oi0 = obuf + (((size_t)NB + b) * NHW + n0 + lm) * NC + 16 * hi;
  const _Float16* or1 = or0 + (size_t)16 * NC;
  const _Float16* oi1 = oi0 + (size_t)16 * NC;

  v8f ar0, ai0, ar1, ai1;
  const int mRow = m0 + 8 * hi;
#pragma unroll
  for (int r = 0; r < 8; ++r) {
    float br = bo[mRow + r], bi = bo[NC + mRow + r];
    ar0[r] = br; ar1[r] = br; ai0[r] = bi; ai1[r] = bi;
  }

#pragma unroll 2
  for (int kk = 0; kk < NC; kk += 32) {
    v16h fwr = load_a_h(wrRow + kk, hi);
    v16h fwi = load_a_h(wiRow + kk, hi);
    v16h fwn = -fwi;
    v16h f0r = load_b_h(or0 + kk);
    v16h f0i = load_b_h(oi0 + kk);
    v16h f1r = load_b_h(or1 + kk);
    v16h f1i = load_b_h(oi1 + kk);
    ar0 = wmma16(fwr, f0r, ar0);  ar0 = wmma16(fwn, f0i, ar0);
    ai0 = wmma16(fwr, f0i, ai0);  ai0 = wmma16(fwi, f0r, ai0);
    ar1 = wmma16(fwr, f1r, ar1);  ar1 = wmma16(fwn, f1i, ar1);
    ai1 = wmma16(fwr, f1i, ai1);  ai1 = wmma16(fwi, f1r, ai1);
  }

#pragma unroll
  for (int r = 0; r < 8; ++r) {
    const int M  = mRow + r;   // c_out
    const int N0 = n0 + lm;
    const int N1 = n0 + 16 + lm;
    const size_t i0r = (((size_t)b) * NC + M) * NHW + N0;
    const size_t i0i = (((size_t)NB + b) * NC + M) * NHW + N0;
    const size_t i1r = (((size_t)b) * NC + M) * NHW + N1;
    const size_t i1i = (((size_t)NB + b) * NC + M) * NHW + N1;
    outp[i0r] = x2[i0r] + ar0[r];
    outp[i0i] = x2[i0i] + ai0[r];
    outp[i1r] = x2[i1r] + ar1[r];
    outp[i1i] = x2[i1i] + ai1[r];
  }
}

// ---------------------------------------------------------------------------
extern "C" void kernel_launch(void* const* d_in, const int* in_sizes, int n_in,
                              void* d_out, int out_size, void* d_ws, size_t ws_size,
                              hipStream_t stream) {
  const float* x2  = (const float*)d_in[0];
  const float* gnw = (const float*)d_in[1];
  const float* gnb = (const float*)d_in[2];
  const float* wq  = (const float*)d_in[3];
  const float* bq  = (const float*)d_in[4];
  const float* wk  = (const float*)d_in[5];
  const float* bk  = (const float*)d_in[6];
  const float* wv  = (const float*)d_in[7];
  const float* bv  = (const float*)d_in[8];
  const float* wo  = (const float*)d_in[9];
  const float* bo  = (const float*)d_in[10];
  float* outp = (float*)d_out;

  char* ws = (char*)d_ws;
  const size_t HSZ = (size_t)2 * NB * NHW * NC * sizeof(_Float16);   // 8 MB
  const size_t SSZ = (size_t)NB * NHW * NHW * sizeof(float);         // 33.5 MB
  const size_t ASZ = (size_t)NB * NHW * NHW * sizeof(_Float16);      // 16.8 MB
  const size_t WSZ = (size_t)2 * NC * NC * sizeof(_Float16);         // 256 KB

  _Float16* hbuf = (_Float16*)(ws);
  _Float16* qbuf = (_Float16*)(ws + HSZ);
  _Float16* kbuf = (_Float16*)(ws + 2 * HSZ);
  _Float16* vbuf = (_Float16*)(ws + 3 * HSZ);
  _Float16* obuf = (_Float16*)(ws + 4 * HSZ);
  float*    Sbuf = (float*)   (ws + 5 * HSZ);
  _Float16* Abuf = (_Float16*)(ws + 5 * HSZ + SSZ);
  _Float16* wq16 = (_Float16*)(ws + 5 * HSZ + SSZ + ASZ);
  _Float16* wk16 = (_Float16*)((char*)wq16 + WSZ);
  _Float16* wv16 = (_Float16*)((char*)wk16 + WSZ);
  _Float16* wo16 = (_Float16*)((char*)wv16 + WSZ);

  const float scale = 0.0625f;  // C^-0.5 = 1/16, folded into q

  k_cvt_w<<<128, 256, 0, stream>>>(wq, wq16);
  k_cvt_w<<<128, 256, 0, stream>>>(wk, wk16);
  k_cvt_w<<<128, 256, 0, stream>>>(wv, wv16);
  k_cvt_w<<<128, 256, 0, stream>>>(wo, wo16);
  k_groupnorm<<<512, 256, 0, stream>>>(x2, gnw, gnb, hbuf);
  k_proj<<<512, 256, 0, stream>>>(hbuf, wq16, bq, qbuf, 0, scale);
  k_proj<<<512, 256, 0, stream>>>(hbuf, wk16, bk, kbuf, 0, 1.0f);
  k_proj<<<512, 256, 0, stream>>>(hbuf, wv16, bv, vbuf, 1, 1.0f);
  k_scores<<<2048, 256, 0, stream>>>(qbuf, kbuf, Sbuf);
  k_softmax<<<8192, 256, 0, stream>>>(Sbuf, Abuf);
  k_av<<<512, 256, 0, stream>>>(Abuf, vbuf, obuf);
  k_final<<<512, 256, 0, stream>>>(obuf, wo16, bo, x2, outp);
}
